// PINModel_65893388255611
// MI455X (gfx1250) — compile-verified, matrix operand order checked by reading, not run on hardware
//
#include <hip/hip_runtime.h>
#include <hip/hip_bf16.h>
#include <math.h>

typedef __attribute__((ext_vector_type(16))) _Float16     v16h;
typedef __attribute__((ext_vector_type(8)))  float        v8f;
typedef __attribute__((ext_vector_type(4)))  unsigned int v4u;

#define B_TOTAL 65536
#define NCONT   6
#define NCAT    6
#define NCATEG  100
#define QF      12     // 6 cont + 6 cat features
#define DF      10     // embedding dim per feature
#define HIDC    20     // per-feature hidden
#define L1      30
#define L2      20
#define NPAIRS  78
#define D0TOK   10

// ---- workspace layout (4-byte units) ----
// [0, 2496)           : pairBias  float [NPAIRS][32]  (tok_p @ sW1[20:30] + sb1, zero-padded)
// [2496, 2496+1280)   : B-matrix fragments uint32 [5 tiles][8 vgpr][32 lanes]
//                       tiles: 0,1 = sW1 (cols 0-15 / 16-31), 2,3 = sW2, 4 = sW3
#define PB_WORDS   (NPAIRS * 32)
#define FRAG_WORDS (5 * 8 * 32)

union AFrag {
    v16h h;
    unsigned int u[8];
    v4u q[2];
};

// Assumed 16-bit B (32x16) per-lane layout, mirror of the documented A layout:
// lane n: column = n&15; khalf = n>>4; VGPR v packs K pair
//   k0(v,khalf) = (v<4 ? 2v : 16 + 2(v-4)) + 8*khalf ; pair {k0 (lo16), k0+1 (hi16)}
__device__ __forceinline__ int b_k0(int v, int khalf) {
    return (v < 4 ? 2 * v : 16 + 2 * (v - 4)) + 8 * khalf;
}

// ------------------- prep: pairBias + packed B fragments -------------------
__global__ void pin_prep(const float* __restrict__ tokens,
                         const float* __restrict__ sW1, const float* __restrict__ sb1,
                         const float* __restrict__ sW2, const float* __restrict__ sW3,
                         float* __restrict__ pb, unsigned int* __restrict__ frag) {
    int gid = blockIdx.x * 256 + threadIdx.x;
    if (gid < PB_WORDS) {
        int p = gid >> 5, o = gid & 31;
        float acc = 0.f;
        if (o < L1) {
            acc = sb1[o];
            for (int i = 0; i < D0TOK; ++i)
                acc += tokens[p * D0TOK + i] * sW1[(2 * DF + i) * L1 + o];
        }
        pb[gid] = acc;
    } else if (gid < PB_WORDS + FRAG_WORDS) {
        int fid   = gid - PB_WORDS;
        int tile  = fid >> 8;
        int v     = (fid >> 5) & 7;
        int n     = fid & 31;
        int col16 = n & 15;
        int khalf = n >> 4;
        int k0    = b_k0(v, khalf);
        float f0 = 0.f, f1 = 0.f;
        if (tile < 2) {                     // sW1 rows 0..19 (E_j|E_k), K padded to 32
            int col = col16 + 16 * tile;
            if (col < L1) {
                if (k0     < 2 * DF) f0 = sW1[k0 * L1 + col];
                if (k0 + 1 < 2 * DF) f1 = sW1[(k0 + 1) * L1 + col];
            }
        } else if (tile < 4) {              // sW2 30x20, K pad 32, N pad 32
            int col = col16 + 16 * (tile - 2);
            if (col < L2) {
                if (k0     < L1) f0 = sW2[k0 * L2 + col];
                if (k0 + 1 < L1) f1 = sW2[(k0 + 1) * L2 + col];
            }
        } else {                            // sW3 20x1 -> column 0 only
            if (col16 == 0) {
                if (k0     < L2) f0 = sW3[k0];
                if (k0 + 1 < L2) f1 = sW3[k0 + 1];
            }
        }
        union { _Float16 h[2]; unsigned int u; } pk;
        pk.h[0] = (_Float16)f0;
        pk.h[1] = (_Float16)f1;
        frag[fid] = pk.u;
    }
}

// ------------------- main fused kernel -------------------
// 1 wave = 16 samples; 8 waves / block; 512 blocks.
#define E_STRIDE_H  132   // f16 per E row (120 used) -> 66 dwords, bank-staggered
#define H_STRIDE_H  40    // f16 per h row (80 bytes, 16B aligned, conflict-free halves)

__global__ __launch_bounds__(256) void pin_main(
    const float* __restrict__ x_cont, const int* __restrict__ x_cat,
    const float* __restrict__ exposure,
    const float* __restrict__ cW1, const float* __restrict__ cb1,
    const float* __restrict__ cW2, const float* __restrict__ cb2,
    const float* __restrict__ cat_tables,
    const float* __restrict__ sb2, const float* __restrict__ sb3,
    const float* __restrict__ out_w, const float* __restrict__ out_b,
    const float* __restrict__ pbias, const unsigned int* __restrict__ frag,
    float* __restrict__ out)
{
    __shared__ __align__(16) _Float16 Ebuf[8][16 * E_STRIDE_H]; // 33792 B
    __shared__ __align__(16) _Float16 Hbuf[8][16 * H_STRIDE_H]; // 10240 B

    const int lane = threadIdx.x & 31;
    const int w    = threadIdx.x >> 5;
    const int base = blockIdx.x * 128 + w * 16;

    _Float16* Ew = &Ebuf[w][0];
    _Float16* Hw = &Hbuf[w][0];

    // ---- phase 1: per-sample embeddings E[16][12][10] into LDS (f16) ----
    #pragma unroll 1
    for (int t = lane; t < 16 * QF; t += 32) {
        int row = t / QF, f = t % QF;
        int b = base + row;
        float ph[DF];
        if (f < NCONT) {
            float x = x_cont[b * NCONT + f];
            #pragma unroll
            for (int d = 0; d < DF; ++d) ph[d] = cb2[f * DF + d];
            #pragma unroll 1
            for (int hh = 0; hh < HIDC; ++hh) {
                float hv = fmaxf(x * cW1[f * HIDC + hh] + cb1[f * HIDC + hh], 0.f);
                #pragma unroll
                for (int d = 0; d < DF; ++d) ph[d] += hv * cW2[(f * HIDC + hh) * DF + d];
            }
        } else {
            int c = f - NCONT;
            int idx = x_cat[b * NCAT + c];
            #pragma unroll
            for (int d = 0; d < DF; ++d) ph[d] = cat_tables[(c * NCATEG + idx) * DF + d];
        }
        #pragma unroll
        for (int d = 0; d < DF; ++d)
            Ew[row * E_STRIDE_H + f * DF + d] = (_Float16)ph[d];
    }
    asm volatile("s_wait_dscnt 0x0" ::: "memory");  // E visible to whole wave (in-order LDS)

    // ---- loop-invariant B fragments in registers (5 tiles x 8 VGPRs) ----
    v16h bW1a, bW1b, bW2a, bW2b, bW3;
    {
        union AFrag t0, t1, t2, t3, t4;
        #pragma unroll
        for (int v = 0; v < 8; ++v) {
            t0.u[v] = frag[0 * 256 + v * 32 + lane];
            t1.u[v] = frag[1 * 256 + v * 32 + lane];
            t2.u[v] = frag[2 * 256 + v * 32 + lane];
            t3.u[v] = frag[3 * 256 + v * 32 + lane];
            t4.u[v] = frag[4 * 256 + v * 32 + lane];
        }
        bW1a = t0.h; bW1b = t1.h; bW2a = t2.h; bW2b = t3.h; bW3 = t4.h;
    }

    const int   colLo = lane & 15;
    const int   row   = lane & 15;                  // A-matrix row owned by this lane
    const float s2a   = sb2[colLo];                 // colLo < 16 <= L2
    const float s2b   = (colLo + 16 < L2) ? sb2[colLo + 16] : 0.f;
    const float s3    = sb3[0];
    const unsigned int* Eu = (const unsigned int*)Ew;

    float eacc[8];
    #pragma unroll
    for (int i = 0; i < 8; ++i) eacc[i] = 0.f;

    int p = 0;
    #pragma unroll 1
    for (int j = 0; j < QF; ++j) {
        #pragma unroll 1
        for (int k = j; k < QF; ++k, ++p) {
            // ---- A1: z rows gathered straight from E (layout per ISA 7.12.2) ----
            union AFrag a1;
            const unsigned int* ej = Eu + row * (E_STRIDE_H / 2) + j * 5;
            const unsigned int* ek = Eu + row * (E_STRIDE_H / 2) + k * 5;
            if (lane < 16) {   // K0..7 = Ej[0..7]; K16..23 = Ek[6..9],0,0,0,0
                a1.u[0] = ej[0]; a1.u[1] = ej[1]; a1.u[2] = ej[2]; a1.u[3] = ej[3];
                a1.u[4] = ek[3]; a1.u[5] = ek[4]; a1.u[6] = 0u;    a1.u[7] = 0u;
            } else {           // K8..15 = Ej[8,9],Ek[0..5]; K24..31 = 0
                a1.u[0] = ej[4]; a1.u[1] = ek[0]; a1.u[2] = ek[1]; a1.u[3] = ek[2];
                a1.u[4] = 0u;    a1.u[5] = 0u;    a1.u[6] = 0u;    a1.u[7] = 0u;
            }

            // ---- layer 1: 2x WMMA, bias (incl. folded token term) in C ----
            float pb0 = pbias[p * 32 + colLo];
            float pb1 = pbias[p * 32 + 16 + colLo];
            v8f c0, c1;
            #pragma unroll
            for (int i = 0; i < 8; ++i) { c0[i] = pb0; c1[i] = pb1; }
            v8f h1a = __builtin_amdgcn_wmma_f32_16x16x32_f16(false, a1.h, false, bW1a, (short)0, c0, false, false);
            v8f h1b = __builtin_amdgcn_wmma_f32_16x16x32_f16(false, a1.h, false, bW1b, (short)0, c1, false, false);

            // relu -> f16 -> LDS rows (transpose C-layout -> A-layout via LDS)
            {
                int r0 = (lane < 16) ? 0 : 8;
                #pragma unroll
                for (int i = 0; i < 8; ++i) {
                    Hw[(r0 + i) * H_STRIDE_H + colLo]      = (_Float16)fmaxf(h1a[i], 0.f);
                    Hw[(r0 + i) * H_STRIDE_H + colLo + 16] = (_Float16)fmaxf(h1b[i], 0.f);
                }
            }
            asm volatile("s_wait_dscnt 0x0" ::: "memory");

            // ---- A2 fragment: two ds_load_b128 per lane ----
            union AFrag a2;
            {
                const v4u* Hq = (const v4u*)(Hw + row * H_STRIDE_H);
                if (lane < 16) { a2.q[0] = Hq[0]; a2.q[1] = Hq[2]; }  // K0..7, K16..23
                else           { a2.q[0] = Hq[1]; a2.q[1] = Hq[3]; }  // K8..15, K24..31
            }

            v8f c2a, c2b;
            #pragma unroll
            for (int i = 0; i < 8; ++i) { c2a[i] = s2a; c2b[i] = s2b; }
            v8f h2a = __builtin_amdgcn_wmma_f32_16x16x32_f16(false, a2.h, false, bW2a, (short)0, c2a, false, false);
            v8f h2b = __builtin_amdgcn_wmma_f32_16x16x32_f16(false, a2.h, false, bW2b, (short)0, c2b, false, false);

            {
                int r0 = (lane < 16) ? 0 : 8;
                #pragma unroll
                for (int i = 0; i < 8; ++i) {
                    Hw[(r0 + i) * H_STRIDE_H + colLo]      = (_Float16)fmaxf(h2a[i], 0.f);
                    Hw[(r0 + i) * H_STRIDE_H + colLo + 16] = (_Float16)fmaxf(h2b[i], 0.f);
                }
            }
            asm volatile("s_wait_dscnt 0x0" ::: "memory");

            // ---- layer 3 (20 -> 1): WMMA against column-0-only B tile ----
            union AFrag a3;
            {
                const v4u* Hq = (const v4u*)(Hw + row * H_STRIDE_H);
                if (lane < 16) { a3.q[0] = Hq[0]; a3.q[1] = Hq[2]; }
                else           { a3.q[0] = Hq[1]; a3.q[1] = Hq[3]; }
            }
            v8f c3;
            #pragma unroll
            for (int i = 0; i < 8; ++i) c3[i] = s3;
            v8f raw = __builtin_amdgcn_wmma_f32_16x16x32_f16(false, a3.h, false, bW3, (short)0, c3, false, false);

            // hard-sigmoid + weighted accumulate (valid in lanes 0 and 16 = col 0)
            float wp = out_w[p];
            #pragma unroll
            for (int i = 0; i < 8; ++i) {
                float ht = fminf(fmaxf(raw[i] * (1.0f / 6.0f) + 0.5f, 0.f), 1.f) - 0.5f;
                eacc[i] += wp * ht;
            }
        }
    }

    // ---- epilogue: lanes 0 / 16 hold rows 0..7 / 8..15 ----
    if (colLo == 0) {
        int r0 = (lane < 16) ? 0 : 8;
        float ob = out_b[0];
        #pragma unroll
        for (int i = 0; i < 8; ++i) {
            int b = base + r0 + i;
            float eta = fminf(fmaxf(eacc[i] + ob, -20.f), 20.f);
            out[b] = __expf(eta) * exposure[b];
        }
    }
}

extern "C" void kernel_launch(void* const* d_in, const int* in_sizes, int n_in,
                              void* d_out, int out_size, void* d_ws, size_t ws_size,
                              hipStream_t stream) {
    (void)in_sizes; (void)n_in; (void)out_size; (void)ws_size;

    const float* x_cont     = (const float*)d_in[0];
    const int*   x_cat      = (const int*)  d_in[1];
    const float* exposure   = (const float*)d_in[2];
    const float* cont_W1    = (const float*)d_in[3];
    const float* cont_b1    = (const float*)d_in[4];
    const float* cont_W2    = (const float*)d_in[5];
    const float* cont_b2    = (const float*)d_in[6];
    const float* cat_tables = (const float*)d_in[7];
    const float* tokens     = (const float*)d_in[8];
    const float* sW1        = (const float*)d_in[9];
    const float* sb1        = (const float*)d_in[10];
    const float* sW2        = (const float*)d_in[11];
    const float* sb2        = (const float*)d_in[12];
    const float* sW3        = (const float*)d_in[13];
    const float* sb3        = (const float*)d_in[14];
    const float* out_w      = (const float*)d_in[15];
    const float* out_b      = (const float*)d_in[16];
    float*       out        = (float*)d_out;

    float*        pb   = (float*)d_ws;                       // PB_WORDS floats
    unsigned int* frag = (unsigned int*)d_ws + PB_WORDS;     // FRAG_WORDS uints (16B aligned: 9984%16==0)

    // prep: 2496 + 1280 = 3776 tasks
    pin_prep<<<dim3((PB_WORDS + FRAG_WORDS + 255) / 256), dim3(256), 0, stream>>>(
        tokens, sW1, sb1, sW2, sW3, pb, frag);

    // main: 512 blocks x 256 threads (8 waves x 16 samples)
    pin_main<<<dim3(B_TOTAL / 128), dim3(256), 0, stream>>>(
        x_cont, x_cat, exposure, cont_W1, cont_b1, cont_W2, cont_b2,
        cat_tables, sb2, sb3, out_w, out_b, pb, frag, out);
}
